// MultiHeadInfiniAttention_61314953118549
// MI455X (gfx1250) — matrix operand order repaired and checked by reading, not compiled, not run on hardware
//
#include <hip/hip_runtime.h>
#include <hip/hip_bf16.h>

// ---------------------------------------------------------------------------
// MultiHeadInfiniAttention on gfx1250 (MI455X), wave32 + WMMA f16->f32 + TDM.
//   B=2, T=4096, D=1024, H=8, Dh=128, SEG=512, S=8
// Kernel 1: qkv projection GEMMs (x @ W + b) -> f16 workspace [b,h,t,d]
// Kernel 2: per-(b,h) segment scan; every contraction via
//           v_wmma_f32_16x16x32_f16; raw tile staging via tensor_load_to_lds
//           (TDM) with LDS row padding, fallback to vectorized copies.
// ---------------------------------------------------------------------------

typedef __attribute__((ext_vector_type(16))) _Float16 v16h;
typedef __attribute__((ext_vector_type(8)))  _Float16 v8h;
typedef __attribute__((ext_vector_type(8)))  float    v8f;
typedef __attribute__((ext_vector_type(4)))  unsigned int v4u;
typedef __attribute__((ext_vector_type(8)))  int      v8i;
typedef __attribute__((ext_vector_type(4)))  int      v4i;

#define EPSV 1e-6f
#define LSTR 136   // LDS row stride in halves (272B: 16B aligned, bank-skewed)

#if __has_builtin(__builtin_amdgcn_tensor_load_to_lds) && \
    __has_builtin(__builtin_amdgcn_s_wait_tensorcnt)
#define HAVE_TDM 1
#endif

// ---- WMMA helper: D = A(16x32 f16) * B(32x16 f16) + C(f32) ----------------
__device__ __forceinline__ v8f wmma32(v16h a, v16h b, v8f c) {
  return __builtin_amdgcn_wmma_f32_16x16x32_f16(
      /*neg_a=*/false, a, /*neg_b=*/false, b,
      /*c_mod=*/(short)0, c, /*reuse_a=*/false, /*reuse_b=*/false);
}

// A fragment (16x32, rows M, K pattern {0..7,16..23}/{8..15,24..31}) from
// LDS array stored [M][K] with row stride `stride` halves.
__device__ __forceinline__ v16h ldA(const _Float16* base, int stride,
                                    int row0, int k0, int lane) {
  int r  = row0 + (lane & 15);
  int kb = k0 + ((lane & 16) ? 8 : 0);
  const _Float16* p = base + r * stride + kb;
  v8h lo = *(const v8h*)(p);
  v8h hi = *(const v8h*)(p + 16);
  return __builtin_shufflevector(lo, hi, 0,1,2,3,4,5,6,7,8,9,10,11,12,13,14,15);
}

// B fragment (32x16, cols N, K pattern {0..7,16..23}/{8..15,24..31}) from LDS array
// stored transposed [N][K] with row stride `stride` halves.
__device__ __forceinline__ v16h ldB(const _Float16* base, int stride,
                                    int col0, int k0, int lane) {
  int n  = col0 + (lane & 15);
  int kb = k0 + ((lane & 16) ? 8 : 0);
  const _Float16* p = base + n * stride + kb;
  v8h lo = *(const v8h*)(p);
  v8h hi = *(const v8h*)(p + 16);
  return __builtin_shufflevector(lo, hi, 0,1,2,3,4,5,6,7,8,9,10,11,12,13,14,15);
}

__device__ __forceinline__ float redmax16(float v) {
  v = fmaxf(v, __shfl_xor(v, 1));
  v = fmaxf(v, __shfl_xor(v, 2));
  v = fmaxf(v, __shfl_xor(v, 4));
  v = fmaxf(v, __shfl_xor(v, 8));
  return v;
}
__device__ __forceinline__ float redsum16(float v) {
  v += __shfl_xor(v, 1);
  v += __shfl_xor(v, 2);
  v += __shfl_xor(v, 4);
  v += __shfl_xor(v, 8);
  return v;
}

#ifdef HAVE_TDM
// TDM: copy a 128x128 f16 tile (global row stride 128 halves) into LDS with
// a 136-half padded row stride. D# built per CDNA5 ISA:
//   g0: count=1 | lds_addr | global_addr | type=2
//   g1: data_size=2B, pad_enable, pad_interval=64 DWORDs, pad_amount=4 DWORDs,
//       tensor_dim0=128, tensor_dim1=128, tile_dim0=128, tile_dim1=128,
//       tensor_dim0_stride=128
__device__ __forceinline__ void tdm_tile_load(unsigned lds_off, const _Float16* gsrc) {
  unsigned long long ga = (unsigned long long)(uintptr_t)gsrc;
  v4u g0;
  g0[0] = 1u;                                            // count=1
  g0[1] = lds_off;                                       // lds_addr (bytes)
  g0[2] = (unsigned)(ga & 0xffffffffu);                  // global_addr[31:0]
  g0[3] = (unsigned)((ga >> 32) & 0x01ffffffu) | (2u << 30); // addr[56:32]|type=2
  v8i g1;
  g1[0] = (int)((1u << 16) | (1u << 20) | (5u << 22) | (3u << 25));
  g1[1] = (int)(128u << 16);   // tensor_dim0=128 (low 16 bits at [31:16])
  g1[2] = (int)(128u << 16);   // tensor_dim0 hi=0; tensor_dim1=128 (low bits)
  g1[3] = (int)(128u << 16);   // tensor_dim1 hi=0; tile_dim0=128
  g1[4] = (int)(128u);         // tile_dim1=128; tile_dim2=0
  g1[5] = (int)(128u);         // tensor_dim0_stride = 128 (low 32)
  g1[6] = 0;                   // stride hi / tensor_dim1_stride low = 0
  g1[7] = 0;
  v4i z4; z4[0] = z4[1] = z4[2] = z4[3] = 0;
#if __clang_major__ >= 23
  v8i z8; for (int i = 0; i < 8; ++i) z8[i] = 0;
  __builtin_amdgcn_tensor_load_to_lds(g0, g1, z4, z4, z8, 0);
#else
  __builtin_amdgcn_tensor_load_to_lds(g0, g1, z4, z4, 0);
#endif
}
#endif

// Fallback raw tile copy (global f16 [128][128] -> LDS [128][LSTR]).
__device__ __forceinline__ void stage_tile(_Float16* dst, const _Float16* src, int tid) {
  for (int i = 0; i < 8; ++i) {
    int idx = i * 256 + tid;
    int l = idx >> 4, c8 = (idx & 15) * 8;
    *(v8h*)&dst[l * LSTR + c8] = *(const v8h*)&src[(size_t)l * 128 + c8];
  }
}

// ---------------------------------------------------------------------------
// Kernel 1: y = x @ W + bias, M=8192, N=1024, K=1024 (z selects q/k/v).
// Output f16, layout out[((b*8+h)*4096 + t)*128 + d].
// ---------------------------------------------------------------------------
__global__ __launch_bounds__(256) void qkv_gemm(
    const float* __restrict__ x,
    const float* __restrict__ w0, const float* __restrict__ w1, const float* __restrict__ w2,
    const float* __restrict__ bb0, const float* __restrict__ bb1, const float* __restrict__ bb2,
    _Float16* __restrict__ o0, _Float16* __restrict__ o1, _Float16* __restrict__ o2)
{
  const int z = blockIdx.z;
  const float* W    = (z == 0) ? w0  : (z == 1) ? w1  : w2;
  const float* bias = (z == 0) ? bb0 : (z == 1) ? bb1 : bb2;
  _Float16*    out  = (z == 0) ? o0  : (z == 1) ? o1  : o2;

  const int tid = threadIdx.x, lane = tid & 31, w = tid >> 5;
  const int wm = w >> 1, wn = w & 1;            // waves: 4 (M) x 2 (N)
  const int m0 = blockIdx.x * 128, n0 = blockIdx.y * 128;
  const int lane15 = lane & 15, hi = lane >> 4;

  __shared__ __align__(16) _Float16 Ah[128 * 40];
  __shared__ __align__(16) _Float16 Bh[128 * 40];

  v8f acc[2][4];
  for (int r = 0; r < 2; ++r)
    for (int c = 0; c < 4; ++c)
      for (int j = 0; j < 8; ++j) acc[r][c][j] = 0.f;

  for (int kt = 0; kt < 32; ++kt) {
    __syncthreads();
    for (int i = 0; i < 16; ++i) {              // A tile 128x32 -> f16
      int idx = i * 256 + tid;
      int l = idx >> 5, kk = idx & 31;
      Ah[l * 40 + kk] = (_Float16)x[(size_t)(m0 + l) * 1024 + kt * 32 + kk];
    }
    for (int i = 0; i < 16; ++i) {              // B tile 32x128 -> Bt[n][k]
      int idx = i * 256 + tid;
      int n = idx & 127, kk = idx >> 7;
      Bh[n * 40 + kk] = (_Float16)W[(size_t)(kt * 32 + kk) * 1024 + n0 + n];
    }
    __syncthreads();
#pragma unroll
    for (int r = 0; r < 2; ++r) {
      v16h a = ldA(Ah, 40, wm * 32 + r * 16, 0, lane);
#pragma unroll
      for (int c = 0; c < 4; ++c) {
        v16h bf = ldB(Bh, 40, wn * 64 + c * 16, 0, lane);
        acc[r][c] = wmma32(a, bf, acc[r][c]);
      }
    }
  }

#pragma unroll
  for (int r = 0; r < 2; ++r)
#pragma unroll
    for (int c = 0; c < 4; ++c)
#pragma unroll
      for (int j = 0; j < 8; ++j) {
        int ml = wm * 32 + r * 16 + j + 8 * hi;
        int nl = wn * 64 + c * 16 + lane15;
        int rm = m0 + ml, nn = n0 + nl;
        int bb = rm >> 12, tt = rm & 4095;
        int hh = nn >> 7,  dd = nn & 127;
        out[(((size_t)bb * 8 + hh) * 4096 + tt) * 128 + dd] =
            (_Float16)(acc[r][c][j] + bias[nn]);
      }
}

// ---------------------------------------------------------------------------
// Kernel 2: per-(b,h) InfiniAttention scan. One block of 256 threads (8 waves)
// per (b,h). M kept in fp32 C-fragment registers; f16 transposed mirror in LDS.
// ---------------------------------------------------------------------------
__global__ __launch_bounds__(256) void infini_attn(
    const _Float16* __restrict__ qg, const _Float16* __restrict__ kg,
    const _Float16* __restrict__ vg, const float* __restrict__ beta,
    float* __restrict__ out)
{
  const int bh = blockIdx.x;
  const int b = bh >> 3, h = bh & 7;
  const int tid = threadIdx.x, lane = tid & 31, w = tid >> 5;
  const int lane15 = lane & 15, hi = lane >> 4;
  const float gate  = 1.f / (1.f + __expf(-beta[h]));
  const float scale = 0.08838834764831845f;     // 1/sqrt(128)

  __shared__ __align__(16) _Float16 MhT[128 * LSTR];   // M^T: [e][d] f16
  __shared__ __align__(16) _Float16 bufQ[128 * LSTR];  // qc[l][d]      ; skT[d][l]
  __shared__ __align__(16) _Float16 bufK[128 * LSTR];  // sq / kc / P   ; sk[l][d]
  __shared__ __align__(16) _Float16 bufV[128 * LSTR];  // vcT[d][m]     ; raw k / deltaT[e][l]
  __shared__ float den_s[128];
  __shared__ float zsh[128];

#ifdef HAVE_TDM
  const unsigned offQ = (unsigned)(uintptr_t)(void*)bufQ;
  const unsigned offK = (unsigned)(uintptr_t)(void*)bufK;
  const unsigned offV = (unsigned)(uintptr_t)(void*)bufV;
#endif

  v8f Macc[8];                                  // persistent M rows d=w*16..+15
#pragma unroll
  for (int c = 0; c < 8; ++c)
#pragma unroll
    for (int j = 0; j < 8; ++j) Macc[c][j] = 0.f;

  for (int i = tid; i < 128 * LSTR; i += 256) MhT[i] = (_Float16)0.f;
  if (tid < 128) zsh[tid] = 0.f;
  __syncthreads();

  const size_t segbase0 = (size_t)bh * 4096 * 128;

  for (int s = 0; s < 8; ++s) {
    // ================= output pass (uses OLD M, OLD z) =================
    for (int qt = 0; qt < 4; ++qt) {
      size_t qbase = segbase0 + (size_t)(s * 512 + qt * 128) * 128;
      __syncthreads();
#ifdef HAVE_TDM
      if (tid < 32) {                            // raw qc -> bufQ via TDM
        tdm_tile_load(offQ, qg + qbase);
        __builtin_amdgcn_s_wait_tensorcnt(0);
      }
#else
      stage_tile(bufQ, qg + qbase, tid);
#endif
      __syncthreads();
      for (int i = 0; i < 8; ++i) {              // sq = elu(qc)+1 -> bufK
        int idx = i * 256 + tid;
        int l = idx >> 4, c8 = (idx & 15) * 8;
        v8h q8 = *(const v8h*)&bufQ[l * LSTR + c8];
        v8h s8;
#pragma unroll
        for (int e = 0; e < 8; ++e) {
          float v = (float)q8[e];
          s8[e] = (_Float16)(v > 0.f ? v + 1.f : __expf(v));
        }
        *(v8h*)&bufK[l * LSTR + c8] = s8;
      }
      __syncthreads();
      if (tid < 128) {                           // den = sq . z_old
        float sum = 0.f;
        for (int d = 0; d < 128; ++d) sum += (float)bufK[tid * LSTR + d] * zsh[d];
        den_s[tid] = sum;
      }
      v8f Am[8];                                 // a_mem numerator = sq @ M
#pragma unroll
      for (int c = 0; c < 8; ++c)
#pragma unroll
        for (int j = 0; j < 8; ++j) Am[c][j] = 0.f;
#pragma unroll
      for (int k = 0; k < 4; ++k) {
        v16h a = ldA(bufK, LSTR, w * 16, k * 32, lane);
#pragma unroll
        for (int c = 0; c < 8; ++c)
          Am[c] = wmma32(a, ldB(MhT, LSTR, c * 16, k * 32, lane), Am[c]);
      }
      __syncthreads();                           // sq reads done; bufK reusable

      // -------- causal flash attention over key tiles kt <= qt --------
      v8f O[8];
      float m_run[8], l_run[8];
#pragma unroll
      for (int j = 0; j < 8; ++j) { m_run[j] = -1e30f; l_run[j] = 0.f; }
#pragma unroll
      for (int c = 0; c < 8; ++c)
#pragma unroll
        for (int j = 0; j < 8; ++j) O[c][j] = 0.f;

      for (int kt = 0; kt <= qt; ++kt) {
        size_t kbase = segbase0 + (size_t)(s * 512 + kt * 128) * 128;
#ifdef HAVE_TDM
        if (tid < 32) tdm_tile_load(offK, kg + kbase);   // kc -> bufK (async)
#else
        stage_tile(bufK, kg + kbase, tid);
#endif
        for (int i = 0; i < 8; ++i) {            // vcT[d][m] (overlaps TDM)
          int idx = i * 256 + tid;
          int l = idx >> 4, c8 = (idx & 15) * 8;
          v8h vv = *(const v8h*)&vg[kbase + (size_t)l * 128 + c8];
#pragma unroll
          for (int e = 0; e < 8; ++e) bufV[(c8 + e) * LSTR + l] = vv[e];
        }
#ifdef HAVE_TDM
        if (tid < 32) __builtin_amdgcn_s_wait_tensorcnt(0);
#endif
        __syncthreads();
        v8f Sv[8];                               // S = qc @ kc^T
#pragma unroll
        for (int c = 0; c < 8; ++c)
#pragma unroll
          for (int j = 0; j < 8; ++j) Sv[c][j] = 0.f;
#pragma unroll
        for (int k = 0; k < 4; ++k) {
          v16h a = ldA(bufQ, LSTR, w * 16, k * 32, lane);
#pragma unroll
          for (int c = 0; c < 8; ++c)
            Sv[c] = wmma32(a, ldB(bufK, LSTR, c * 16, k * 32, lane), Sv[c]);
        }
        const bool diag = (kt == qt);
#pragma unroll
        for (int c = 0; c < 8; ++c) {
          int col = kt * 128 + c * 16 + lane15;
#pragma unroll
          for (int j = 0; j < 8; ++j) {
            int row = qt * 128 + w * 16 + j + 8 * hi;
            float sv = Sv[c][j] * scale;
            Sv[c][j] = (!diag || col <= row) ? sv : -1e30f;
          }
        }
#pragma unroll
        for (int j = 0; j < 8; ++j) {            // online softmax per row
          float mx = -3e38f;
#pragma unroll
          for (int c = 0; c < 8; ++c) mx = fmaxf(mx, Sv[c][j]);
          mx = redmax16(mx);
          float mnew  = fmaxf(m_run[j], mx);
          float alpha = __expf(m_run[j] - mnew);
          float rs = 0.f;
#pragma unroll
          for (int c = 0; c < 8; ++c) { float p = __expf(Sv[c][j] - mnew); Sv[c][j] = p; rs += p; }
          rs = redsum16(rs);
          m_run[j] = mnew;
          l_run[j] = l_run[j] * alpha + rs;
#pragma unroll
          for (int c = 0; c < 8; ++c) O[c][j] *= alpha;
        }
        __syncthreads();                         // all done reading bufK as K
#pragma unroll
        for (int c = 0; c < 8; ++c)              // P -> LDS (reuse bufK)
#pragma unroll
          for (int j = 0; j < 8; ++j)
            bufK[(w * 16 + j + 8 * hi) * LSTR + c * 16 + lane15] = (_Float16)Sv[c][j];
        __syncthreads();
#pragma unroll
        for (int k = 0; k < 4; ++k) {            // O += P @ vc
          v16h a = ldA(bufK, LSTR, w * 16, k * 32, lane);
#pragma unroll
          for (int c = 0; c < 8; ++c)
            O[c] = wmma32(a, ldB(bufV, LSTR, c * 16, k * 32, lane), O[c]);
        }
        __syncthreads();
      }

      // combine: out = gate * a_mem/(den+eps) + (1-gate) * O/l
#pragma unroll
      for (int c = 0; c < 8; ++c)
#pragma unroll
        for (int j = 0; j < 8; ++j) {
          int rl = w * 16 + j + 8 * hi;
          float am = Am[c][j] / (den_s[rl] + EPSV);
          float ad = O[c][j] / l_run[j];
          int t = s * 512 + qt * 128 + rl;
          out[((size_t)b * 4096 + t) * 1024 + h * 128 + c * 16 + lane15] =
              gate * am + (1.f - gate) * ad;
        }
    } // qt

    // ================= memory update pass =================
    float zacc = 0.f;
    for (int rt = 0; rt < 4; ++rt) {
      size_t kbase = segbase0 + (size_t)(s * 512 + rt * 128) * 128;
      __syncthreads();
#ifdef HAVE_TDM
      if (tid < 32) {                            // raw kc -> bufV via TDM
        tdm_tile_load(offV, kg + kbase);
        __builtin_amdgcn_s_wait_tensorcnt(0);
      }
#else
      stage_tile(bufV, kg + kbase, tid);
#endif
      __syncthreads();
      for (int i = 0; i < 8; ++i) {              // sk -> bufK[l][d], bufQ[d][l]
        int idx = i * 256 + tid;
        int l = idx >> 4, c8 = (idx & 15) * 8;
        v8h k8 = *(const v8h*)&bufV[l * LSTR + c8];
        v8h s8;
#pragma unroll
        for (int e = 0; e < 8; ++e) {
          float v = (float)k8[e];
          s8[e] = (_Float16)(v > 0.f ? v + 1.f : __expf(v));
        }
        *(v8h*)&bufK[l * LSTR + c8] = s8;
#pragma unroll
        for (int e = 0; e < 8; ++e) bufQ[(c8 + e) * LSTR + l] = s8[e];
      }
      __syncthreads();
      if (tid < 128) {
        float sum = 0.f;                         // denk = sk . z_old
        for (int d = 0; d < 128; ++d) sum += (float)bufK[tid * LSTR + d] * zsh[d];
        den_s[tid] = sum;
        float cs = 0.f;                          // column-sum for z update
        for (int l2 = 0; l2 < 128; ++l2) cs += (float)bufK[l2 * LSTR + tid];
        zacc += cs;
      }
      v8f R[8];                                  // retr = sk @ M_old
#pragma unroll
      for (int c = 0; c < 8; ++c)
#pragma unroll
        for (int j = 0; j < 8; ++j) R[c][j] = 0.f;
#pragma unroll
      for (int k = 0; k < 4; ++k) {
        v16h a = ldA(bufK, LSTR, w * 16, k * 32, lane);
#pragma unroll
        for (int c = 0; c < 8; ++c)
          R[c] = wmma32(a, ldB(MhT, LSTR, c * 16, k * 32, lane), R[c]);
      }
      __syncthreads();
#pragma unroll
      for (int c = 0; c < 8; ++c)                // deltaT[e][l] = v - retr/den
#pragma unroll
        for (int j = 0; j < 8; ++j) {
          int rl = w * 16 + j + 8 * hi;
          float vv = (float)vg[kbase + (size_t)rl * 128 + c * 16 + lane15];
          float dl = vv - R[c][j] / (den_s[rl] + EPSV);
          bufV[(c * 16 + lane15) * LSTR + rl] = (_Float16)dl;
        }
      __syncthreads();
#pragma unroll
      for (int k = 0; k < 4; ++k) {              // M += sk^T @ delta
        v16h a = ldA(bufQ, LSTR, w * 16, k * 32, lane);
#pragma unroll
        for (int c = 0; c < 8; ++c)
          Macc[c] = wmma32(a, ldB(bufV, LSTR, c * 16, k * 32, lane), Macc[c]);
      }
    } // rt
    __syncthreads();
#pragma unroll
    for (int c = 0; c < 8; ++c)                  // refresh f16 mirror MhT[e][d]
#pragma unroll
      for (int j = 0; j < 8; ++j)
        MhT[(c * 16 + lane15) * LSTR + w * 16 + j + 8 * hi] = (_Float16)Macc[c][j];
    if (tid < 128) zsh[tid] += zacc;
    __syncthreads();
  } // s
}

// ---------------------------------------------------------------------------
extern "C" void kernel_launch(void* const* d_in, const int* in_sizes, int n_in,
                              void* d_out, int out_size, void* d_ws, size_t ws_size,
                              hipStream_t stream) {
  (void)in_sizes; (void)n_in; (void)out_size; (void)ws_size;
  const float* x    = (const float*)d_in[0];
  const float* w_q  = (const float*)d_in[1];
  const float* b_q  = (const float*)d_in[2];
  const float* w_k  = (const float*)d_in[3];
  const float* b_k  = (const float*)d_in[4];
  const float* w_v  = (const float*)d_in[5];
  const float* b_v  = (const float*)d_in[6];
  const float* beta = (const float*)d_in[7];
  float* out = (float*)d_out;

  _Float16* qw = (_Float16*)d_ws;               // [16][4096][128] f16
  _Float16* kw = qw + (size_t)8192 * 1024;
  _Float16* vw = kw + (size_t)8192 * 1024;

  dim3 g1(64, 8, 3);
  qkv_gemm<<<g1, 256, 0, stream>>>(x, w_q, w_k, w_v, b_q, b_k, b_v, qw, kw, vw);
  infini_attn<<<16, 256, 0, stream>>>(qw, kw, vw, beta, out);
}
